// CFConvFused_5042291605796
// MI455X (gfx1250) — compile-verified, matrix operand order checked by reading, not run on hardware
//
#include <hip/hip_runtime.h>
#include <math.h>

typedef __attribute__((ext_vector_type(16))) _Float16 v16h;
typedef __attribute__((ext_vector_type(8)))  float    v8f;

#define N_NODES  100000
#define N_EDGES  1600000
#define D_RADIAL 50
#define D_HID    64

// ---------------------------------------------------------------------------
// f16 WMMA operand fragment from an f32 row-major matrix.
// Lane l (g = l>>4) holds halves i=0..7  -> K = kbase + 8g + i,
//                         halves i=8..15 -> K = kbase + 16 + 8g + (i-8).
// K >= kmax zero-filled (K-padding for D_RADIAL=50 -> 64).
// ---------------------------------------------------------------------------
__device__ __forceinline__ v16h load_frag_f32(const float* __restrict__ mat,
                                              int row, int ld, int kbase, int kmax) {
  const int l  = threadIdx.x & 31;
  const int g8 = (l >> 4) << 3;   // 0 or 8
  const float* rp = mat + (size_t)row * (size_t)ld;
  v16h r;
#pragma unroll
  for (int i = 0; i < 16; ++i) {
    const int k = kbase + ((i < 8) ? (g8 + i) : (16 + g8 + (i - 8)));
    const float v = (k < kmax) ? rp[k] : 0.0f;
    r[i] = (_Float16)v;
  }
  return r;
}

__device__ __forceinline__ v8f wmma_f16(v16h a, v16h b, v8f c) {
  return __builtin_amdgcn_wmma_f32_16x16x32_f16(false, a, false, b,
                                                (short)0, c, false, false);
}

// ---------------------------------------------------------------------------
// Node GEMM: out[m,n] = act( sum_k in[m,k]*W[n,k] + bias[n] )
// Wave pair: waves (2p, 2p+1) share a 16-row tile; wave handles 32 channels
// (2 N-tiles). B fragments stay register-resident (32 VGPRs) -> no spills.
// ---------------------------------------------------------------------------
template <int SILU>
__global__ void node_gemm_kernel(const float* __restrict__ in,
                                 const float* __restrict__ W,
                                 const float* __restrict__ bias,
                                 float* __restrict__ out,
                                 int nrows) {
  const int l    = threadIdx.x & 31;
  const int n    = l & 15;
  const int g8   = (l >> 4) << 3;
  const int wave = (int)((blockIdx.x * blockDim.x + threadIdx.x) >> 5);
  const int half = wave & 1;            // which 32 output channels
  const int pair = wave >> 1;
  const int npairs = (int)((gridDim.x * blockDim.x) >> 6);
  const int nb0  = half * 2;            // first N-tile index (0 or 2)

  v16h Bf[2][2];
  float bv[2];
#pragma unroll
  for (int j = 0; j < 2; ++j) {
#pragma unroll
    for (int kt = 0; kt < 2; ++kt)
      Bf[j][kt] = load_frag_f32(W, (nb0 + j) * 16 + n, 64, kt * 32, 64);
    bv[j] = bias[(nb0 + j) * 16 + n];
  }

  const int ntiles = nrows >> 4;
  for (int t = pair; t < ntiles; t += npairs) {
    const int m0 = t << 4;
    const v16h A0 = load_frag_f32(in, m0 + n, 64, 0, 64);
    const v16h A1 = load_frag_f32(in, m0 + n, 64, 32, 64);

#pragma unroll
    for (int j = 0; j < 2; ++j) {
      v8f acc;
#pragma unroll
      for (int r = 0; r < 8; ++r) acc[r] = bv[j];
      acc = wmma_f16(A0, Bf[j][0], acc);
      acc = wmma_f16(A1, Bf[j][1], acc);
#pragma unroll
      for (int r = 0; r < 8; ++r) {
        float z = acc[r];
        if (SILU)  // compile-time; fast sigmoid: v_exp_f32 + v_rcp_f32
          z = z * __builtin_amdgcn_rcpf(1.0f + __expf(-z));
        out[(size_t)(m0 + r + g8) * D_HID + (nb0 + j) * 16 + n] = z;
      }
    }
  }
}

// ---------------------------------------------------------------------------
// Fused edge pass: filt = eb @ W_rad.T + b_rad (f16 WMMA, f32 acc),
//                  msg = hv[src] * filt,  h[dst] += msg (f32 atomics, L2-resident).
// Wave pair shares a 16-edge tile (A + index loads hit WGP$); each wave owns
// 32 channels. Gathers are issued BEFORE the WMMAs so matrix latency hides
// them, and atomics go out as a batch instead of load->wait->atomic chains.
// ---------------------------------------------------------------------------
__global__ void edge_kernel(const float* __restrict__ eb,
                            const int*   __restrict__ src,
                            const int*   __restrict__ dst,
                            const float* __restrict__ W_rad,
                            const float* __restrict__ b_rad,
                            const float* __restrict__ hv,
                            float* __restrict__ h) {
  const int l    = threadIdx.x & 31;
  const int n    = l & 15;
  const int g8   = (l >> 4) << 3;
  const int wave = (int)((blockIdx.x * blockDim.x + threadIdx.x) >> 5);
  const int half = wave & 1;
  const int pair = wave >> 1;
  const int npairs = (int)((gridDim.x * blockDim.x) >> 6);
  const int nb0  = half * 2;

  v16h Bf[2][2];
  float bv[2];
#pragma unroll
  for (int j = 0; j < 2; ++j) {
#pragma unroll
    for (int kt = 0; kt < 2; ++kt)
      Bf[j][kt] = load_frag_f32(W_rad, (nb0 + j) * 16 + n, D_RADIAL, kt * 32, D_RADIAL);
    bv[j] = b_rad[(nb0 + j) * 16 + n];
  }

  const int ntiles = N_EDGES >> 4;     // 100000 tiles of 16 edges
  for (int t = pair; t < ntiles; t += npairs) {
    const int m0 = t << 4;

    // Edge indices -> 32-bit element offsets (node*64 + n); scale_offset does *4.
    int soff[8], doff[8];
#pragma unroll
    for (int r = 0; r < 8; ++r) {
      soff[r] = src[m0 + r + g8] * D_HID + n;
      doff[r] = dst[m0 + r + g8] * D_HID + n;
    }

    // Prefetch all 16 gathered source-node values (overlaps with WMMAs below).
    float gv[2][8];
#pragma unroll
    for (int j = 0; j < 2; ++j)
#pragma unroll
      for (int r = 0; r < 8; ++r)
        gv[j][r] = hv[soff[r] + (nb0 + j) * 16];

    const v16h A0 = load_frag_f32(eb, m0 + n, D_RADIAL, 0,  D_RADIAL);
    const v16h A1 = load_frag_f32(eb, m0 + n, D_RADIAL, 32, D_RADIAL);

#pragma unroll
    for (int j = 0; j < 2; ++j) {
      v8f acc;
#pragma unroll
      for (int r = 0; r < 8; ++r) acc[r] = bv[j];
      acc = wmma_f16(A0, Bf[j][0], acc);   // K = 0..31
      acc = wmma_f16(A1, Bf[j][1], acc);   // K = 32..49 (50..63 zero)
#pragma unroll
      for (int r = 0; r < 8; ++r)
        atomicAdd(&h[doff[r] + (nb0 + j) * 16], acc[r] * gv[j][r]);
    }
  }
}

extern "C" void kernel_launch(void* const* d_in, const int* in_sizes, int n_in,
                              void* d_out, int out_size, void* d_ws, size_t ws_size,
                              hipStream_t stream) {
  const float* x      = (const float*)d_in[0];
  const float* eb     = (const float*)d_in[1];
  const int*   src    = (const int*)  d_in[2];
  const int*   dst    = (const int*)  d_in[3];
  const float* W_pre  = (const float*)d_in[4];
  const float* b_pre  = (const float*)d_in[5];
  const float* W_rad  = (const float*)d_in[6];
  const float* b_rad  = (const float*)d_in[7];
  const float* W_post = (const float*)d_in[8];
  const float* b_post = (const float*)d_in[9];
  float* out = (float*)d_out;

  float* hv = (float*)d_ws;                              // N_NODES * 64 f32
  float* h  = hv + (size_t)N_NODES * D_HID;              // N_NODES * 64 f32

  hipMemsetAsync(h, 0, (size_t)N_NODES * D_HID * sizeof(float), stream);

  const dim3 blk(256);  // 8 waves = 4 wave-pairs per block (wave32)
  node_gemm_kernel<0><<<dim3(512),  blk, 0, stream>>>(x, W_pre,  b_pre,  hv, N_NODES);
  edge_kernel       <<<dim3(2048), blk, 0, stream>>>(eb, src, dst, W_rad, b_rad, hv, h);
  node_gemm_kernel<1><<<dim3(512),  blk, 0, stream>>>(h, W_post, b_post, out, N_NODES);
}
